// EdgeBlock_69853348102604
// MI455X (gfx1250) — matrix operand order repaired and checked by reading, not compiled, not run on hardware
//
#include <hip/hip_runtime.h>
#include <hip/hip_bf16.h>

typedef __attribute__((ext_vector_type(16))) __bf16 v16bf;
typedef __attribute__((ext_vector_type(8)))  float  v8f;

#define FD      128           // feature dim D
#define TILE_E  64            // edges per workgroup
#define LDSPAD  8             // pad (bf16 elems) to break bank conflicts
#define ASTR    (FD + LDSPAD) // 136 elems -> 272B row stride (16B aligned)

// sched_group_barrier masks
#define SG_WMMA   0x008
#define SG_DSREAD 0x100

__device__ __forceinline__ unsigned short f2bf(float f) {
    // round-to-nearest-even fp32 -> bf16 (bit pattern)
    unsigned int u = __float_as_uint(f);
    u += 0x7FFFu + ((u >> 16) & 1u);
    return (unsigned short)(u >> 16);
}

union Frag { v16bf v; uint4 q[2]; };

__global__ __launch_bounds__(256)
void edgeblock_wmma_bf16(const float* __restrict__ edges,
                         const float* __restrict__ nodes,
                         const float* __restrict__ glob,
                         const int*   __restrict__ recv,
                         const int*   __restrict__ send,
                         const float* __restrict__ W,
                         const float* __restrict__ bias,
                         float* __restrict__ out,
                         int E)
{
    // bf16 staging buffers (raw ushort bit patterns)
    __shared__ __align__(16) unsigned short sA[TILE_E][ASTR]; // 64 x 128 A-chunk
    __shared__ __align__(16) unsigned short sW[FD][ASTR];     // 128 x 128 W-chunk

    const int tid  = threadIdx.x;       // 0..255
    const int lane = tid & 31;
    const int wave = tid >> 5;          // 0..7
    const long long e0 = (long long)blockIdx.x * TILE_E;

    // wave -> output stripe: 16 edges (m0) x 64 features (n0base..+63)
    const int m0     = (wave & 3) * 16;
    const int n0base = (wave >> 2) * 64;

    // fragment addressing (invariant across K-steps)
    const int arow = m0 + (lane & 15);
    const int kbo  = (lane >> 4) << 3;                    // A: K base 0/8 by lane half
    const int kro  = (lane & 15) + ((lane >> 4) << 4);    // B: K row by lane

    v8f acc[4] = {};          // 4 N-tiles of 16x16 f32 accumulators
    Frag af[2], bf[2][4];     // double-buffered fragment registers

    for (int kc = 0; kc < 4; ++kc) {        // K chunk == concat segment
        if (kc) __syncthreads();

        // ---- stage A chunk: 64 edges x 128 feats of segment kc -> bf16 LDS ----
        for (int i = 0; i < 8; ++i) {
            int er = wave + i * 8;                       // edge row 0..63
            long long e = e0 + er; if (e >= (long long)E) e = (long long)E - 1;
            const float* src;
            if      (kc == 0) src = edges + e * FD;
            else if (kc == 1) src = nodes + (long long)recv[e] * FD;
            else if (kc == 2) src = nodes + (long long)send[e] * FD;
            else              src = glob;
            float4 f = *(const float4*)(src + lane * 4);
            ushort4 h = make_ushort4(f2bf(f.x), f2bf(f.y), f2bf(f.z), f2bf(f.w));
            *(ushort4*)&sA[er][lane * 4] = h;
        }

        // ---- stage W chunk: rows kc*128..+127, all 128 cols -> bf16 LDS ----
        for (int j = 0; j < 8; ++j) {
            int idx = tid + j * 256;            // 0..2047 chunks of 8 floats
            int row = idx >> 4;                 // 0..127
            int c8  = (idx & 15) * 8;           // 0..120
            const float* wsrc = W + (long long)(kc * FD + row) * FD + c8;
            float4 f0 = *(const float4*)(wsrc);
            float4 f1 = *(const float4*)(wsrc + 4);
            union { ushort4 h[2]; uint4 q; } p;
            p.h[0] = make_ushort4(f2bf(f0.x), f2bf(f0.y), f2bf(f0.z), f2bf(f0.w));
            p.h[1] = make_ushort4(f2bf(f1.x), f2bf(f1.y), f2bf(f1.z), f2bf(f1.w));
            *(uint4*)&sW[row][c8] = p.q;
        }

        __syncthreads();

        // ---- GEMM on chunk: 4 K-steps of 32, software-pipelined fragments ----
        auto load_frags = [&](int ks, int buf) {
            int kb = ks * 32 + kbo;
            af[buf].q[0] = *(const uint4*)&sA[arow][kb];
            af[buf].q[1] = *(const uint4*)&sA[arow][kb + 16];
            int krow = ks * 32 + kro;
            #pragma unroll
            for (int t = 0; t < 4; ++t) {
                int n0 = n0base + t * 16;
                unsigned short* bp = (unsigned short*)&bf[buf][t];
                #pragma unroll
                for (int s = 0; s < 8; ++s) {
                    bp[s]     = sW[kb + s][n0 + (lane & 15)];
                    bp[8 + s] = sW[kb + 16 + s][n0 + (lane & 15)];
                }
            }
        };

        load_frags(0, 0);
        // pin the initial 10 fragment loads as the first DS-read group
        __builtin_amdgcn_sched_group_barrier(SG_DSREAD, 10, 0);

        #pragma unroll
        for (int ks = 0; ks < 4; ++ks) {
            int cur = ks & 1;
            if (ks < 3) {
                load_frags(ks + 1, cur ^ 1);   // prefetch next K-step
            }
            #pragma unroll
            for (int t = 0; t < 4; ++t) {      // 4 independent WMMAs
                acc[t] = __builtin_amdgcn_wmma_f32_16x16x32_bf16(
                            false, af[cur].v, false, bf[cur][t].v,
                            (short)0, acc[t], false, false);
            }
            // enforce pipeline order: [next-step DS loads] then [this step's WMMAs]
            if (ks < 3) __builtin_amdgcn_sched_group_barrier(SG_DSREAD, 10, 0);
            __builtin_amdgcn_sched_group_barrier(SG_WMMA, 4, 0);
        }
    }

    // ---- epilogue: bias + ReLU + store (C/D layout: VGPR i -> M=i / M=8+i) ----
    #pragma unroll
    for (int t = 0; t < 4; ++t) {
        int n = n0base + t * 16 + (lane & 15);
        float bv = bias[n];
        long long rbase = e0 + m0 + ((lane >> 4) << 3);
        #pragma unroll
        for (int i = 0; i < 8; ++i) {
            long long row = rbase + i;
            if (row < (long long)E) {
                float v = acc[t][i] + bv;
                out[row * FD + n] = v > 0.0f ? v : 0.0f;
            }
        }
    }
}

extern "C" void kernel_launch(void* const* d_in, const int* in_sizes, int n_in,
                              void* d_out, int out_size, void* d_ws, size_t ws_size,
                              hipStream_t stream) {
    const float* edges = (const float*)d_in[0];
    const float* nodes = (const float*)d_in[1];
    const float* glob  = (const float*)d_in[2];
    const int*   recv  = (const int*)d_in[3];
    const int*   send  = (const int*)d_in[4];
    const float* W     = (const float*)d_in[5];
    const float* bias  = (const float*)d_in[6];
    float* out = (float*)d_out;

    int E = in_sizes[0] / FD;                 // 800000
    int grid = (E + TILE_E - 1) / TILE_E;     // 12500 workgroups
    hipLaunchKernelGGL(edgeblock_wmma_bf16, dim3(grid), dim3(256), 0, stream,
                       edges, nodes, glob, recv, send, W, bias, out, E);
}
